// TimedPoincareTransformerLayer_69621419868919
// MI455X (gfx1250) — matrix operand order repaired and checked by reading, not compiled, not run on hardware
//
#include <hip/hip_runtime.h>
#include <hip/hip_bf16.h>
#include <cmath>

#define DD   1024
#define HH   16
#define HDD  64
#define FFD  4096
#define BB   4
#define SS   1024
#define MTOT (BB * SS)  // 4096 tokens

typedef __bf16 bf16_t;
typedef __attribute__((ext_vector_type(16))) __bf16 v16bf;
typedef __attribute__((ext_vector_type(8)))  __bf16 v8bf;
typedef __attribute__((ext_vector_type(8)))  float  v8f;

// ---------- helpers ----------

__device__ __forceinline__ unsigned short f2bfbits(float f) {
  unsigned u = __float_as_uint(f);
  return (unsigned short)((u + 0x7FFFu + ((u >> 16) & 1u)) >> 16);
}

// Assemble a 16-element bf16 fragment from two 16-byte chunks.
__device__ __forceinline__ v16bf ld_frag(const bf16_t* p0, const bf16_t* p1) {
  v8bf lo = *(const v8bf*)p0;
  v8bf hi = *(const v8bf*)p1;
  v16bf r;
#pragma unroll
  for (int i = 0; i < 8; ++i) { r[i] = lo[i]; r[i + 8] = hi[i]; }
  return r;
}

__device__ __forceinline__ v8f wmma_bf16(v16bf a, v16bf b, v8f c) {
  return __builtin_amdgcn_wmma_f32_16x16x32_bf16(false, a, false, b, (short)0, c,
                                                 false, false);
}

// 256-thread (8 wave32) block reduction; s must hold 8 floats.
__device__ __forceinline__ float block_reduce_sum(float v, float* s) {
  __syncthreads();
#pragma unroll
  for (int off = 16; off > 0; off >>= 1) v += __shfl_xor(v, off, 32);
  if ((threadIdx.x & 31) == 0) s[threadIdx.x >> 5] = v;
  __syncthreads();
  float r = 0.f;
#pragma unroll
  for (int i = 0; i < 8; ++i) r += s[i];
  return r;
}

// ---------- weight convert: fp32 [K,N] -> bf16 [N,K], LDS-tiled transpose ----------
__global__ __launch_bounds__(256) void k_wconv(const float* __restrict__ W,
                                               unsigned short* __restrict__ Wt,
                                               int K, int N) {
  __shared__ unsigned short tile[32][33];  // pad to kill bank conflicts
  int kb = blockIdx.x * 32;
  int nb = blockIdx.y * 32;
  int tx = threadIdx.x & 31;   // fast index
  int ty = threadIdx.x >> 5;   // 8 row-groups
#pragma unroll
  for (int i = 0; i < 32; i += 8) {
    int k = kb + ty + i;
    tile[ty + i][tx] = f2bfbits(W[(size_t)k * N + nb + tx]);  // coalesced over n
  }
  __syncthreads();
#pragma unroll
  for (int i = 0; i < 32; i += 8) {
    int n = nb + ty + i;
    Wt[(size_t)n * K + kb + tx] = tile[tx][ty + i];           // coalesced over k
  }
}

// ---------- adaLN: ada[b,j] = t_emb[b,:] @ w_ada[:,j] + b_ada[j] (tiny, f32) ----------
__global__ __launch_bounds__(256) void k_ada(const float* __restrict__ temb,
                                             const float* __restrict__ w,
                                             const float* __restrict__ bvec,
                                             float* __restrict__ ada) {
  int idx = blockIdx.x * 256 + threadIdx.x;
  int b = idx / (6 * DD);
  int j = idx % (6 * DD);
  const float* t = temb + (size_t)b * DD;
  float acc = bvec[j];
  for (int k = 0; k < DD; ++k) acc = fmaf(t[k], w[(size_t)k * (6 * DD) + j], acc);
  ada[idx] = acc;
}

// ---------- fused logmap0 + LayerNorm + adaLN modulation -> bf16 row ----------
__global__ __launch_bounds__(256) void k_prep(const float* __restrict__ X,
                                              const float* __restrict__ ada,
                                              int sh_off, int sc_off,
                                              unsigned short* __restrict__ U) {
  __shared__ float red[8];
  int tok = blockIdx.x;
  int b = tok >> 10;
  const float* xr = X + (size_t)tok * DD;
  const float* ar = ada + (size_t)b * (6 * DD);
  float xl[4];
  float sx = 0.f, sxx = 0.f;
#pragma unroll
  for (int i = 0; i < 4; ++i) {
    float v = xr[threadIdx.x + 256 * i];
    xl[i] = v; sx += v; sxx += v * v;
  }
  sx  = block_reduce_sum(sx, red);
  sxx = block_reduce_sum(sxx, red);
  float n   = sqrtf(sxx);
  float nc  = fmaxf(n, 1e-7f);                       // _norm clip
  float arg = fminf(nc, 1.0f - 1e-5f);               // tanh-arg clip
  float lsc = atanhf(arg) / nc;                      // logmap0 scale
  float m   = lsc * sx * (1.0f / DD);
  float var = lsc * lsc * sxx * (1.0f / DD) - m * m;
  float inv = rsqrtf(var + 1e-6f);
#pragma unroll
  for (int i = 0; i < 4; ++i) {
    int c = threadIdx.x + 256 * i;
    float xn = (lsc * xl[i] - m) * inv;
    float u = xn * (1.0f + ar[sc_off + c]) + ar[sh_off + c];
    U[(size_t)tok * DD + c] = f2bfbits(u);
  }
}

// ---------- bf16 WMMA GEMM: out = alpha * (A[M,K] @ Wt[N,K]^T + bias) ----------
// Wave tile 32x64 (2 A-frags reuse 4 B-frags -> 8 WMMA / K-step).
// Block tile 128x128 (8 waves: 4 M x 2 N).
// modes: 0 f32 [M,N], 1 bf16 [M,N], 2 bf16 scatter [B,H,S,HD], 3 bf16 scatter [B,H,HD,S]
__global__ __launch_bounds__(256) void k_gemm(const unsigned short* __restrict__ Abits,
                                              int lda,
                                              const unsigned short* __restrict__ Wbits,
                                              const float* __restrict__ bias,
                                              float alpha, int K, int N, int mode,
                                              void* __restrict__ outp) {
  const bf16_t* A  = (const bf16_t*)Abits;
  const bf16_t* Wt = (const bf16_t*)Wbits;
  int lane = threadIdx.x & 31;
  int wave = threadIdx.x >> 5;
  int l16 = lane & 15, hf = lane >> 4;
  int mbase = blockIdx.x * 128 + (wave >> 1) * 32;
  int nbase = blockIdx.y * 128 + (wave & 1) * 64;

  v8f acc[2][4];
#pragma unroll
  for (int rt = 0; rt < 2; ++rt)
#pragma unroll
    for (int ch = 0; ch < 4; ++ch) acc[rt][ch] = {};

  const bf16_t* arow0 = A + (size_t)(mbase + l16) * lda;
  const bf16_t* arow1 = A + (size_t)(mbase + 16 + l16) * lda;
  const bf16_t* w0 = Wt + (size_t)(nbase + 0 * 16 + l16) * K;
  const bf16_t* w1 = Wt + (size_t)(nbase + 1 * 16 + l16) * K;
  const bf16_t* w2 = Wt + (size_t)(nbase + 2 * 16 + l16) * K;
  const bf16_t* w3 = Wt + (size_t)(nbase + 3 * 16 + l16) * K;

  for (int kk = 0; kk < K; kk += 32) {
    int kb = kk + hf * 16;
    v16bf fa0 = ld_frag(arow0 + kk + hf * 8, arow0 + kk + 16 + hf * 8);
    v16bf fa1 = ld_frag(arow1 + kk + hf * 8, arow1 + kk + 16 + hf * 8);
    v16bf fb0 = ld_frag(w0 + kk + hf * 8, w0 + kk + 16 + hf * 8);
    v16bf fb1 = ld_frag(w1 + kk + hf * 8, w1 + kk + 16 + hf * 8);
    v16bf fb2 = ld_frag(w2 + kk + hf * 8, w2 + kk + 16 + hf * 8);
    v16bf fb3 = ld_frag(w3 + kk + hf * 8, w3 + kk + 16 + hf * 8);
    if (kk + 64 < K) {  // prefetch next cacheline on each stream (global_prefetch_b8)
      __builtin_prefetch(arow0 + kk + 64, 0, 1);
      __builtin_prefetch(arow1 + kk + 64, 0, 1);
      __builtin_prefetch(w0 + kb + 64, 0, 1);
      __builtin_prefetch(w1 + kb + 64, 0, 1);
      __builtin_prefetch(w2 + kb + 64, 0, 1);
      __builtin_prefetch(w3 + kb + 64, 0, 1);
    }
    acc[0][0] = wmma_bf16(fa0, fb0, acc[0][0]);
    acc[0][1] = wmma_bf16(fa0, fb1, acc[0][1]);
    acc[0][2] = wmma_bf16(fa0, fb2, acc[0][2]);
    acc[0][3] = wmma_bf16(fa0, fb3, acc[0][3]);
    acc[1][0] = wmma_bf16(fa1, fb0, acc[1][0]);
    acc[1][1] = wmma_bf16(fa1, fb1, acc[1][1]);
    acc[1][2] = wmma_bf16(fa1, fb2, acc[1][2]);
    acc[1][3] = wmma_bf16(fa1, fb3, acc[1][3]);
  }

#pragma unroll
  for (int rt = 0; rt < 2; ++rt) {
#pragma unroll
    for (int ch = 0; ch < 4; ++ch) {
      int col = nbase + ch * 16 + l16;
      float bv = bias ? bias[col] : 0.0f;
#pragma unroll
      for (int r = 0; r < 8; ++r) {
        int row = mbase + rt * 16 + hf * 8 + r;
        float val = alpha * (acc[rt][ch][r] + bv);
        if (mode == 0) {
          ((float*)outp)[(size_t)row * N + col] = val;
        } else if (mode == 1) {
          ((unsigned short*)outp)[(size_t)row * N + col] = f2bfbits(val);
        } else if (mode == 2) {  // [B,H,S,HD]
          size_t idx = (((size_t)((row >> 10) * HH + (col >> 6)) * SS + (row & 1023)) * HDD +
                        (col & 63));
          ((unsigned short*)outp)[idx] = f2bfbits(val);
        } else {                 // [B,H,HD,S]  (V transposed for flash-attn PV)
          size_t idx = (((size_t)((row >> 10) * HH + (col >> 6)) * HDD + (col & 63)) * SS +
                        (row & 1023));
          ((unsigned short*)outp)[idx] = f2bfbits(val);
        }
      }
    }
  }
}

// ---------- flash attention: one wave per (b,h, 16 q rows) ----------
__global__ __launch_bounds__(32) void k_attn(const unsigned short* __restrict__ Qb,
                                             const unsigned short* __restrict__ Kb,
                                             const unsigned short* __restrict__ VTb,
                                             unsigned short* __restrict__ AO) {
  const bf16_t* Q  = (const bf16_t*)Qb;
  const bf16_t* Km = (const bf16_t*)Kb;
  const bf16_t* VT = (const bf16_t*)VTb;
  int lane = threadIdx.x;
  int l16 = lane & 15, hf = lane >> 4;
  int qt = blockIdx.x & (SS / 16 - 1);
  int bh = blockIdx.x >> 6;  // / (SS/16)
  int b = bh >> 4, h = bh & 15;
  const bf16_t* qh = Q  + (size_t)bh * SS * HDD;
  const bf16_t* kh = Km + (size_t)bh * SS * HDD;
  const bf16_t* vh = VT + (size_t)bh * HDD * SS;
  int q0 = qt * 16;

  const bf16_t* qrow = qh + (size_t)(q0 + l16) * HDD;
  v16bf aq0 = ld_frag(qrow + hf * 8,      qrow + 16 + hf * 8);   // d 0..31
  v16bf aq1 = ld_frag(qrow + 32 + hf * 8, qrow + 48 + hf * 8);   // d 32..63

  v8f o0 = {}, o1 = {}, o2 = {}, o3 = {};
  float rmax[8], rsum[8];
#pragma unroll
  for (int r = 0; r < 8; ++r) { rmax[r] = -1e30f; rsum[r] = 0.f; }

  __shared__ unsigned short pl[16 * 32];  // P tile relayout buffer (1KB)

  for (int kb = 0; kb < SS; kb += 32) {
    // ---- scores for 32 keys: two 16x16 WMMA chains over K(d)=64 ----
    const bf16_t* kr0 = kh + (size_t)(kb + l16) * HDD;
    const bf16_t* kr1 = kh + (size_t)(kb + 16 + l16) * HDD;
    v16bf b00 = ld_frag(kr0 + hf * 8,      kr0 + 16 + hf * 8);
    v16bf b01 = ld_frag(kr0 + 32 + hf * 8, kr0 + 48 + hf * 8);
    v16bf b10 = ld_frag(kr1 + hf * 8,      kr1 + 16 + hf * 8);
    v16bf b11 = ld_frag(kr1 + 32 + hf * 8, kr1 + 48 + hf * 8);
    v8f c0 = {}, c1 = {};
    c0 = wmma_bf16(aq0, b00, c0);
    c0 = wmma_bf16(aq1, b01, c0);
    c1 = wmma_bf16(aq0, b10, c1);
    c1 = wmma_bf16(aq1, b11, c1);

    // ---- online softmax (rows live as half*8+r, cols across 16 lanes) ----
    float nm[8], al[8];
#pragma unroll
    for (int r = 0; r < 8; ++r) {
      float mx = fmaxf(c0[r], c1[r]);
#pragma unroll
      for (int off = 1; off < 16; off <<= 1) mx = fmaxf(mx, __shfl_xor(mx, off, 32));
      float newm = fmaxf(rmax[r], mx);
      al[r] = __expf(rmax[r] - newm);
      nm[r] = newm;
      rmax[r] = newm;
    }
#pragma unroll
    for (int r = 0; r < 8; ++r) {
      float p0 = __expf(c0[r] - nm[r]);
      float p1 = __expf(c1[r] - nm[r]);
      pl[(hf * 8 + r) * 32 + l16]      = f2bfbits(p0);
      pl[(hf * 8 + r) * 32 + 16 + l16] = f2bfbits(p1);
      float s = p0 + p1;
#pragma unroll
      for (int off = 1; off < 16; off <<= 1) s += __shfl_xor(s, off, 32);
      rsum[r] = rsum[r] * al[r] + s;
      o0[r] *= al[r]; o1[r] *= al[r]; o2[r] *= al[r]; o3[r] *= al[r];
    }
    __syncthreads();
    // ---- P (16x32) C-layout -> A-fragment via LDS ----
    const bf16_t* prow = (const bf16_t*)pl + l16 * 32;
    v16bf pf = ld_frag(prow + hf * 16, prow + hf * 16 + 8);
    __syncthreads();
    // ---- PV: B frags from V^T (contiguous over keys) ----
    const bf16_t* v0r = vh + (size_t)(0 * 16 + l16) * SS + kb + hf * 16;
    const bf16_t* v1r = vh + (size_t)(1 * 16 + l16) * SS + kb + hf * 16;
    const bf16_t* v2r = vh + (size_t)(2 * 16 + l16) * SS + kb + hf * 16;
    const bf16_t* v3r = vh + (size_t)(3 * 16 + l16) * SS + kb + hf * 16;
    o0 = wmma_bf16(pf, ld_frag(v0r, v0r + 8), o0);
    o1 = wmma_bf16(pf, ld_frag(v1r, v1r + 8), o1);
    o2 = wmma_bf16(pf, ld_frag(v2r, v2r + 8), o2);
    o3 = wmma_bf16(pf, ld_frag(v3r, v3r + 8), o3);
  }

  // normalize + store concat-head layout [B,S,D] bf16
  size_t base = ((size_t)(b * SS + q0)) * DD + (size_t)h * HDD;
#pragma unroll
  for (int r = 0; r < 8; ++r) {
    float inv = 1.0f / rsum[r];
    size_t rowoff = base + (size_t)(hf * 8 + r) * DD;
    AO[rowoff + 0 * 16 + l16] = f2bfbits(o0[r] * inv);
    AO[rowoff + 1 * 16 + l16] = f2bfbits(o1[r] * inv);
    AO[rowoff + 2 * 16 + l16] = f2bfbits(o2[r] * inv);
    AO[rowoff + 3 * 16 + l16] = f2bfbits(o3[r] * inv);
  }
}

// ---------- fused gate + expmap0 + mobius_add residual ----------
__global__ __launch_bounds__(256) void k_residual(const float* __restrict__ X,
                                                  const float* __restrict__ pre,
                                                  const float* __restrict__ ada,
                                                  int g_off, float* __restrict__ out) {
  __shared__ float red[8];
  int tok = blockIdx.x;
  int b = tok >> 10;
  const float* xr = X + (size_t)tok * DD;
  const float* pr = pre + (size_t)tok * DD;
  const float* ar = ada + (size_t)b * (6 * DD) + g_off;
  float xl[4], tl[4];
  float sx2 = 0.f, st2 = 0.f, sxt = 0.f;
#pragma unroll
  for (int i = 0; i < 4; ++i) {
    int c = threadIdx.x + 256 * i;
    float x = xr[c];
    float t = ar[c] * pr[c];
    xl[i] = x; tl[i] = t;
    sx2 += x * x; st2 += t * t; sxt += x * t;
  }
  sx2 = block_reduce_sum(sx2, red);
  st2 = block_reduce_sum(st2, red);
  sxt = block_reduce_sum(sxt, red);
  float nt  = fmaxf(sqrtf(st2), 1e-7f);
  float tn  = tanhf(nt);
  float esc = tn / nt;                 // expmap0 scale: e = esc * t
  float e2  = esc * esc * st2;
  float xy  = esc * sxt;
  float ca  = 1.0f + 2.0f * xy + e2;   // coef of x
  float cb  = 1.0f - sx2;              // coef of e
  float den = fmaxf(1.0f + 2.0f * xy + sx2 * e2, 1e-7f);
  float rinv = 1.0f / den;
#pragma unroll
  for (int i = 0; i < 4; ++i) {
    int c = threadIdx.x + 256 * i;
    out[(size_t)tok * DD + c] = (ca * xl[i] + cb * esc * tl[i]) * rinv;
  }
}

// ---------- host ----------
extern "C" void kernel_launch(void* const* d_in, const int* in_sizes, int n_in,
                              void* d_out, int out_size, void* d_ws, size_t ws_size,
                              hipStream_t stream) {
  (void)in_sizes; (void)n_in; (void)out_size; (void)ws_size;
  const float* x     = (const float*)d_in[0];
  const float* temb  = (const float*)d_in[1];
  const float* w_q   = (const float*)d_in[2];
  const float* b_q   = (const float*)d_in[3];
  const float* w_k   = (const float*)d_in[4];
  const float* b_k   = (const float*)d_in[5];
  const float* w_v   = (const float*)d_in[6];
  const float* b_v   = (const float*)d_in[7];
  const float* w_o   = (const float*)d_in[8];
  const float* b_o   = (const float*)d_in[9];
  const float* w_f1  = (const float*)d_in[10];
  const float* b_f1  = (const float*)d_in[11];
  const float* w_f2  = (const float*)d_in[12];
  const float* b_f2  = (const float*)d_in[13];
  const float* w_ada = (const float*)d_in[14];
  const float* b_ada = (const float*)d_in[15];

  char* ws = (char*)d_ws;
  size_t off = 0;
  auto alloc = [&](size_t bytes) -> void* {
    off = (off + 255) & ~(size_t)255;
    void* p = ws + off;
    off += bytes;
    return p;
  };

  float*          ada  = (float*)alloc(sizeof(float) * BB * 6 * DD);
  unsigned short* WtQ  = (unsigned short*)alloc(2ull * DD * DD);
  unsigned short* WtK  = (unsigned short*)alloc(2ull * DD * DD);
  unsigned short* WtV  = (unsigned short*)alloc(2ull * DD * DD);
  unsigned short* WtO  = (unsigned short*)alloc(2ull * DD * DD);
  unsigned short* WtF1 = (unsigned short*)alloc(2ull * FFD * DD);
  unsigned short* WtF2 = (unsigned short*)alloc(2ull * DD * FFD);
  unsigned short* U    = (unsigned short*)alloc(2ull * MTOT * DD);
  unsigned short* Qs   = (unsigned short*)alloc(2ull * MTOT * DD);
  unsigned short* Ks   = (unsigned short*)alloc(2ull * MTOT * DD);
  unsigned short* VTs  = (unsigned short*)alloc(2ull * MTOT * DD);
  unsigned short* AO   = (unsigned short*)alloc(2ull * MTOT * DD);
  float*          OPRE = (float*)alloc(4ull * MTOT * DD);
  float*          Y    = (float*)alloc(4ull * MTOT * DD);
  unsigned short* H1   = (unsigned short*)alloc(2ull * MTOT * FFD);
  float*          H2   = (float*)alloc(4ull * MTOT * DD);

  // beta-split scale: scores = (r^2/sqrt(HD)) * (a_q . a_k), folded into q
  double BN = exp(lgamma(512.0) + lgamma(0.5) - lgamma(512.5));
  double BH = exp(lgamma(32.0)  + lgamma(0.5) - lgamma(32.5));
  double r  = BH / BN;
  float alphaQ = (float)(r * r / 8.0);

  dim3 b256(256);
  // weight conversion (fp32 [K,N] -> bf16 [N,K]), LDS-tiled
  k_wconv<<<dim3(DD / 32,  DD / 32),  b256, 0, stream>>>(w_q,  WtQ,  DD,  DD);
  k_wconv<<<dim3(DD / 32,  DD / 32),  b256, 0, stream>>>(w_k,  WtK,  DD,  DD);
  k_wconv<<<dim3(DD / 32,  DD / 32),  b256, 0, stream>>>(w_v,  WtV,  DD,  DD);
  k_wconv<<<dim3(DD / 32,  DD / 32),  b256, 0, stream>>>(w_o,  WtO,  DD,  DD);
  k_wconv<<<dim3(DD / 32,  FFD / 32), b256, 0, stream>>>(w_f1, WtF1, DD,  FFD);
  k_wconv<<<dim3(FFD / 32, DD / 32),  b256, 0, stream>>>(w_f2, WtF2, FFD, DD);

  k_ada<<<(BB * 6 * DD) / 256, b256, 0, stream>>>(temb, w_ada, b_ada, ada);

  // --- attention block ---
  k_prep<<<MTOT, b256, 0, stream>>>(x, ada, 0 * DD, 1 * DD, U);
  dim3 g1(MTOT / 128, DD / 128);
  k_gemm<<<g1, b256, 0, stream>>>(U, DD, WtQ, b_q, alphaQ, DD, DD, 2, Qs);
  k_gemm<<<g1, b256, 0, stream>>>(U, DD, WtK, b_k, 1.0f,   DD, DD, 2, Ks);
  k_gemm<<<g1, b256, 0, stream>>>(U, DD, WtV, b_v, 1.0f,   DD, DD, 3, VTs);
  k_attn<<<BB * HH * (SS / 16), dim3(32), 0, stream>>>(Qs, Ks, VTs, AO);
  k_gemm<<<g1, b256, 0, stream>>>(AO, DD, WtO, b_o, 1.0f, DD, DD, 0, OPRE);
  k_residual<<<MTOT, b256, 0, stream>>>(x, OPRE, ada, 2 * DD, Y);

  // --- FFN block ---
  k_prep<<<MTOT, b256, 0, stream>>>(Y, ada, 3 * DD, 4 * DD, U);
  dim3 g2(MTOT / 128, FFD / 128);
  k_gemm<<<g2, b256, 0, stream>>>(U, DD, WtF1, b_f1, 1.0f, DD, FFD, 1, H1);
  k_gemm<<<g1, b256, 0, stream>>>(H1, FFD, WtF2, b_f2, 1.0f, FFD, DD, 0, H2);
  k_residual<<<MTOT, b256, 0, stream>>>(Y, H2, ada, 5 * DD, (float*)d_out);
}
